// SelfAttentionLayerSparse_19842748907978
// MI455X (gfx1250) — matrix operand order repaired and checked by reading, not compiled, not run on hardware
//
#include <hip/hip_runtime.h>
#include <hip/hip_bf16.h>

#define NNODES 50000
#define NEDGES 800000
#define FIN 128
#define FQK 128
#define FTOT 384          // 2*FQK + FV
#define NHEAD 8

typedef __attribute__((ext_vector_type(16))) _Float16 v16h;
typedef __attribute__((ext_vector_type(8)))  _Float16 v8h;
typedef __attribute__((ext_vector_type(8)))  float    v8f;

// ---------------- helpers ----------------

// monotonic float <-> uint encoding so unsigned atomicMax == float max
__device__ __forceinline__ unsigned fenc(float f) {
    unsigned u = __float_as_uint(f);
    return (u & 0x80000000u) ? ~u : (u | 0x80000000u);
}
__device__ __forceinline__ float fdec(unsigned u) {
    return __uint_as_float((u & 0x80000000u) ? (u ^ 0x80000000u) : ~u);
}

__device__ __forceinline__ int lower_bound(const int* __restrict__ a, int n, int key) {
    int lo = 0, hi = n;
    while (lo < hi) {
        int mid = (lo + hi) >> 1;
        if (a[mid] < key) lo = mid + 1; else hi = mid;
    }
    return lo;
}

// ---------------- kernel 0: init global per-head max ----------------
__global__ void init_gmax_kernel(unsigned* __restrict__ gmax) {
    if (threadIdx.x < NHEAD) gmax[threadIdx.x] = 0u;   // 0 == below enc(-inf)
}

// ---------------- kernel 1: one-shot f32 -> (f16 hi, f16 lo) split ---------
// x: nx elements; W: nw elements with q-scaling (rows < FQK get *0.25) folded.
__global__ void __launch_bounds__(256)
split_kernel(const float* __restrict__ x, const float* __restrict__ W,
             _Float16* __restrict__ xhi, _Float16* __restrict__ xlo,
             _Float16* __restrict__ whi, _Float16* __restrict__ wlo,
             int nx, int nw) {
    const int i = blockIdx.x * blockDim.x + threadIdx.x;
    if (i < nx) {
        const float f = x[i];
        const _Float16 h = (_Float16)f;
        xhi[i] = h;
        xlo[i] = (_Float16)(f - (float)h);
    } else {
        const int j = i - nx;
        if (j < nw) {
            const int wrow = j / FIN;                     // W row = output column
            const float s  = (wrow < FQK) ? 0.25f : 1.0f; // q scaling Fh^-0.5
            const float f  = W[j] * s;
            const _Float16 h = (_Float16)f;
            whi[j] = h;
            wlo[j] = (_Float16)(f - (float)h);
        }
    }
}

// ---------------- kernel 2: qkv = x @ W^T via WMMA (pre-split f16) ---------
// One wave per 16x16 output tile. M=50000 (=3125*16), N=384 (=24*16), K=128.
// Fragments are loaded as raw f16 pairs; inner loop = loads + wmma only.
__global__ void __launch_bounds__(256)
qkv_gemm_wmma(const _Float16* __restrict__ xhi, const _Float16* __restrict__ xlo,
              const _Float16* __restrict__ whi, const _Float16* __restrict__ wlo,
              float* __restrict__ qkv, int n) {
    const int lane = threadIdx.x & 31;
    const int tile = blockIdx.x * 8 + (threadIdx.x >> 5);
    const int TILES_N = FTOT / 16;                // 24
    const int tm = tile / TILES_N;                // consecutive waves share tm -> A hits
    const int tn = tile % TILES_N;
    if (tm * 16 >= n) return;

    const int  l15    = lane & 15;
    const bool hiHalf = lane >= 16;

    const int arow = tm * 16 + l15;               // A: row for this lane
    const int ncol = tn * 16 + l15;               // B: output column (W row)
    const size_t aoff = (size_t)arow * FIN + (hiHalf ? 16 : 0);
    const size_t boff = (size_t)ncol * FIN + (hiHalf ? 16 : 0);

    v8f acc = {};

#pragma unroll
    for (int kb = 0; kb < FIN; kb += 32) {
        // ---- A fragment (16x32 f16): lane<16 holds K=kb+0..15 of row l15,
        //                              lane>=16 holds K=kb+16..31 (contiguous, matches B)
        const _Float16* aph = xhi + aoff + kb;
        const _Float16* apl = xlo + aoff + kb;
        const v8h ah0 = *(const v8h*)(aph);
        const v8h ah1 = *(const v8h*)(aph + 8);
        const v8h al0 = *(const v8h*)(apl);
        const v8h al1 = *(const v8h*)(apl + 8);
        const v16h ahi = __builtin_shufflevector(ah0, ah1,
                0, 1, 2, 3, 4, 5, 6, 7, 8, 9, 10, 11, 12, 13, 14, 15);
        const v16h alo = __builtin_shufflevector(al0, al1,
                0, 1, 2, 3, 4, 5, 6, 7, 8, 9, 10, 11, 12, 13, 14, 15);

        // ---- B fragment (32x16): lane<16 holds K=kb+0..15 of column l15,
        //                          lane>=16 holds K=kb+16..31 (contiguous 16 halves)
        const _Float16* bph = whi + boff + kb;
        const _Float16* bpl = wlo + boff + kb;
        const v8h bh0 = *(const v8h*)(bph);
        const v8h bh1 = *(const v8h*)(bph + 8);
        const v8h bl0 = *(const v8h*)(bpl);
        const v8h bl1 = *(const v8h*)(bpl + 8);
        const v16h bhi = __builtin_shufflevector(bh0, bh1,
                0, 1, 2, 3, 4, 5, 6, 7, 8, 9, 10, 11, 12, 13, 14, 15);
        const v16h blo = __builtin_shufflevector(bl0, bl1,
                0, 1, 2, 3, 4, 5, 6, 7, 8, 9, 10, 11, 12, 13, 14, 15);

        // 3-product f16 split: hi*hi + hi*lo + lo*hi ~= fp32 product
        acc = __builtin_amdgcn_wmma_f32_16x16x32_f16(false, ahi, false, bhi,
                                                     (short)0, acc, false, false);
        acc = __builtin_amdgcn_wmma_f32_16x16x32_f16(false, ahi, false, blo,
                                                     (short)0, acc, false, false);
        acc = __builtin_amdgcn_wmma_f32_16x16x32_f16(false, alo, false, bhi,
                                                     (short)0, acc, false, false);
    }

    // C/D layout: lane<16 -> rows M=0..7 (vgpr i), lane>=16 -> rows M=8..15
    const int mbase = tm * 16 + (hiHalf ? 8 : 0);
    float* op = qkv + (size_t)mbase * FTOT + tn * 16 + l15;
#pragma unroll
    for (int i = 0; i < 8; ++i) op[(size_t)i * FTOT] = acc[i];
}

// ---------------- kernel 3: per-edge scores + global per-head max ----------
// One wave per source node (src is sorted). Lane l covers features 4l..4l+3,
// head h = l>>2. q row stays in registers for the whole edge segment.
__global__ void __launch_bounds__(256)
edge_scores_kernel(const float* __restrict__ qkv, const int* __restrict__ src,
                   const int* __restrict__ dst, float* __restrict__ scores,
                   unsigned* __restrict__ gmax, int n, int E) {
    __shared__ unsigned smax[NHEAD];
    const int lane = threadIdx.x & 31;
    const int node = blockIdx.x * 8 + (threadIdx.x >> 5);
    if (threadIdx.x < NHEAD) smax[threadIdx.x] = 0u;
    __syncthreads();

    const int h = lane >> 2;
    unsigned lmax = 0u;
    if (node < n) {
        const int e0 = lower_bound(src, E, node);
        const int e1 = lower_bound(src, E, node + 1);
        const float4 q4 = *(const float4*)(qkv + (size_t)node * FTOT + lane * 4);
        for (int e = e0; e < e1; ++e) {
            const int d = dst[e];
            if (e + 1 < e1) {   // cover L2 gather latency for the next k row
                const int dn = dst[e + 1];
                __builtin_prefetch(qkv + (size_t)dn * FTOT + FQK + lane * 4, 0, 3);
            }
            const float4 k4 = *(const float4*)(qkv + (size_t)d * FTOT + FQK + lane * 4);
            float p = q4.x * k4.x + q4.y * k4.y + q4.z * k4.z + q4.w * k4.w;
            p += __shfl_xor(p, 1);
            p += __shfl_xor(p, 2);        // all 4 lanes of the head group hold the dot
            if ((lane & 3) == 0) scores[(size_t)e * NHEAD + h] = p;
            const unsigned ep = fenc(p);
            lmax = lmax > ep ? lmax : ep;
        }
        if ((lane & 3) == 0) atomicMax(&smax[h], lmax);   // LDS reduction
    }
    __syncthreads();
    if (threadIdx.x < NHEAD) atomicMax(&gmax[threadIdx.x], smax[threadIdx.x]);
}

// ---------------- kernel 4: softmax denom + weighted V aggregation --------
__global__ void __launch_bounds__(256)
node_aggregate_kernel(const float* __restrict__ qkv, const int* __restrict__ src,
                      const int* __restrict__ dst, const float* __restrict__ scores,
                      const unsigned* __restrict__ gmax, float* __restrict__ out,
                      int n, int E) {
    const int lane = threadIdx.x & 31;
    const int node = blockIdx.x * 8 + (threadIdx.x >> 5);
    if (node >= n) return;
    const int h  = lane >> 2;
    const float gm = fdec(gmax[h]);

    const int e0 = lower_bound(src, E, node);
    const int e1 = lower_bound(src, E, node + 1);

    float denom = 0.f;
    for (int e = e0; e < e1; ++e)
        denom += __expf(scores[(size_t)e * NHEAD + h] - gm);
    const float inv = (denom > 0.f) ? (1.0f / denom) : 0.f;

    float4 acc = make_float4(0.f, 0.f, 0.f, 0.f);
    for (int e = e0; e < e1; ++e) {
        const float a = __expf(scores[(size_t)e * NHEAD + h] - gm) * inv;
        const int d = dst[e];
        if (e + 1 < e1) {   // prefetch next v row
            const int dn = dst[e + 1];
            __builtin_prefetch(qkv + (size_t)dn * FTOT + 2 * FQK + lane * 4, 0, 3);
        }
        const float4 v4 = *(const float4*)(qkv + (size_t)d * FTOT + 2 * FQK + lane * 4);
        acc.x += a * v4.x; acc.y += a * v4.y; acc.z += a * v4.z; acc.w += a * v4.w;
    }
    *(float4*)(out + (size_t)node * 128 + lane * 4) = acc;   // no-edge nodes write 0
}

// ---------------- launch ----------------
extern "C" void kernel_launch(void* const* d_in, const int* in_sizes, int n_in,
                              void* d_out, int out_size, void* d_ws, size_t ws_size,
                              hipStream_t stream) {
    const float* x  = (const float*)d_in[0];
    const float* W  = (const float*)d_in[1];
    // d_in[2] = batch (unused by the reference computation)
    const int*   ei = (const int*)d_in[3];

    const int n  = in_sizes[0] / FIN;      // 50000
    const int nx = in_sizes[0];            // n * 128
    const int nw = in_sizes[1];            // 384 * 128
    const int E  = in_sizes[3] / 2;        // 800000
    const int* src = ei;
    const int* dst = ei + E;

    // workspace layout:
    //   qkv    [n*384]  f32
    //   scores [E*8]    f32
    //   gmax   [16]     u32 (8 used, padded for alignment)
    //   xhi/xlo[n*128]  f16 each
    //   whi/wlo[384*128]f16 each
    float*    qkv    = (float*)d_ws;
    float*    scores = qkv + (size_t)n * FTOT;
    unsigned* gmax   = (unsigned*)(scores + (size_t)E * NHEAD);
    _Float16* xhi    = (_Float16*)(gmax + 16);
    _Float16* xlo    = xhi + (size_t)nx;
    _Float16* whi    = xlo + (size_t)nx;
    _Float16* wlo    = whi + (size_t)nw;
    float*    out    = (float*)d_out;

    init_gmax_kernel<<<1, 32, 0, stream>>>(gmax);

    const int splitElems = nx + nw;
    split_kernel<<<(splitElems + 255) / 256, 256, 0, stream>>>(
        x, W, xhi, xlo, whi, wlo, nx, nw);

    const int tiles = ((n + 15) / 16) * (FTOT / 16);       // 3125 * 24 = 75000
    qkv_gemm_wmma<<<(tiles + 7) / 8, 256, 0, stream>>>(xhi, xlo, whi, wlo, qkv, n);

    const int nodeBlocks = (n + 7) / 8;                    // 8 waves (nodes) / block
    edge_scores_kernel<<<nodeBlocks, 256, 0, stream>>>(qkv, src, dst, scores, gmax, n, E);
    node_aggregate_kernel<<<nodeBlocks, 256, 0, stream>>>(qkv, src, dst, scores, gmax, out, n, E);
}
